// EncoderBlock_89773406421661
// MI455X (gfx1250) — matrix operand order repaired and checked by reading, not compiled, not run on hardware
//
#include <hip/hip_runtime.h>
#include <cstdint>
#include <cstddef>

// ---------------------------------------------------------------- constants
#define B_    2
#define S_    2048
#define DM_   1024
#define H_    16
#define DF_   4096
#define HD_   64
#define NTOK_ (B_ * S_)          // 4096 rows
#define EPS_  1e-6f
#define NEG_  -1e9f

typedef _Float16 f16;
typedef __attribute__((ext_vector_type(16))) _Float16 v16h;
typedef __attribute__((ext_vector_type(8)))  float    v8f;
typedef __attribute__((ext_vector_type(4)))  unsigned int u32x4;

union Frag16 {                    // 16 halves = 8 VGPRs, loaded as 2x b128
    v16h  v;
    u32x4 q[2];
};

static __device__ __forceinline__ v8f wmma_f16(const v16h& a, const v16h& b, v8f c) {
    return __builtin_amdgcn_wmma_f32_16x16x32_f16(false, a, false, b, (short)0, c,
                                                  false, false);
}

// async global->LDS 16B copy (CDNA5 ASYNCcnt path; no VGPR data return)
static __device__ __forceinline__ void async_copy_b128(const f16* g, f16* l) {
    unsigned lds_off = (unsigned)(uintptr_t)l;   // low 32b of generic = LDS offset
    asm volatile("global_load_async_to_lds_b128 %0, %1, off"
                 :: "v"(lds_off), "v"(g) : "memory");
}

// ---------------------------------------------------------------- weight prep
// W is f32 row-major [K][N]; emit f16 transposed [N][K] (K contiguous).
__global__ __launch_bounds__(256)
void wprep_kernel(const float* __restrict__ W, f16* __restrict__ Wt, int K, int N) {
    size_t idx = (size_t)blockIdx.x * 256 + threadIdx.x;
    if (idx >= (size_t)K * N) return;
    int n = (int)(idx / K);
    int k = (int)(idx % K);
    Wt[idx] = (f16)W[(size_t)k * N + n];
}

// ---------------------------------------------------------------- layernorm
// One block per row.  torch-style: mean, unbiased std (ddof=1), eps on std.
__global__ __launch_bounds__(256)
void layernorm_f16_kernel(const float* __restrict__ X,
                          const float* __restrict__ alpha,
                          const float* __restrict__ beta,
                          f16* __restrict__ out) {
    constexpr int D = DM_;
    int row = blockIdx.x;
    int tid = threadIdx.x;
    const float* xr = X + (size_t)row * D;

    float v[4];
    float s = 0.f, ss = 0.f;
#pragma unroll
    for (int i = 0; i < 4; ++i) {
        v[i] = xr[tid + i * 256];
        s += v[i];
        ss += v[i] * v[i];
    }
#pragma unroll
    for (int off = 16; off > 0; off >>= 1) {
        s  += __shfl_down(s,  off, 32);
        ss += __shfl_down(ss, off, 32);
    }
    __shared__ float sb[8], qb[8];
    int w = tid >> 5, lane = tid & 31;
    if (lane == 0) { sb[w] = s; qb[w] = ss; }
    __syncthreads();
    if (tid == 0) {
        float ts = 0.f, tq = 0.f;
#pragma unroll
        for (int k = 0; k < 8; ++k) { ts += sb[k]; tq += qb[k]; }
        sb[0] = ts; qb[0] = tq;
    }
    __syncthreads();
    s = sb[0]; ss = qb[0];

    float mean = s / (float)D;
    float var  = fmaxf((ss - s * mean) / (float)(D - 1), 0.f);
    float inv  = 1.f / (sqrtf(var) + EPS_);
    float a = alpha[0], b = beta[0];
#pragma unroll
    for (int i = 0; i < 4; ++i)
        out[(size_t)row * D + tid + i * 256] = (f16)(a * (v[i] - mean) * inv + b);
}

// ---------------------------------------------------------------- WMMA GEMM
// C[M][N] = A[M][K] (f16 row-major) x B (f16, given transposed: Bt[N][K]).
// 256 threads = 8 waves; block tile 128x256x32, wave tile 64x64 (4x4 frags,
// 16 v_wmma per K-step).  Double-buffered LDS fed by async global->LDS
// copies (ASYNCcnt), overlapping tile k+1 copy with tile k WMMA.
// Epilogue modes:
//   0: f16, head layout  out[((b*H+h)*S+s)*HD+d],  val=(acc+bias)*scale
//   2: f16, V transposed out[((b*H+h)*HD+d)*S+s]
//   3: f16, relu, row-major [M][N]
//   4: f32, row-major, + resid[M][N]
__global__ __launch_bounds__(256)
void gemm_wmma_kernel(const f16* __restrict__ A, const f16* __restrict__ Bt,
                      const float* __restrict__ bias,
                      const float* __restrict__ resid,
                      float* __restrict__ outF, f16* __restrict__ outH,
                      int M, int N, int K, float scale, int mode) {
    constexpr int BM = 128, BN = 256, BK = 32, LDT = 40; // 40 halves = 80B stride
    __shared__ __align__(16) f16 As[2][BM][LDT];   // 20 KB
    __shared__ __align__(16) f16 Bs[2][BN][LDT];   // 40 KB  (WGP has 320 KB)

    int tid  = threadIdx.x;
    int w    = tid >> 5;
    int lane = tid & 31;
    int g    = lane >> 4;     // half-wave group
    int nlo  = lane & 15;

    int ntiles = N / BN;
    int m0 = (blockIdx.x / ntiles) * BM;
    int n0 = (blockIdx.x % ntiles) * BN;

    int mbase = (w >> 2) * 64;  // 0 or 64
    int nbase = (w & 3) * 64;   // 0,64,128,192

    v8f acc[4][4];
#pragma unroll
    for (int i = 0; i < 4; ++i)
#pragma unroll
        for (int j = 0; j < 4; ++j) acc[i][j] = {};

    int ksteps = K / BK;

    // issue async copies for K-tile kt into LDS buffer `buf`
    auto issue_tile = [&](int kt, int buf) {
        int k0 = kt * BK;
#pragma unroll
        for (int j = 0; j < 2; ++j) {        // A: 512 16B segments
            int seg = tid * 2 + j;
            int row = seg >> 2;
            int off = (seg & 3) * 8;
            async_copy_b128(A + (size_t)(m0 + row) * K + k0 + off,
                            &As[buf][row][off]);
        }
#pragma unroll
        for (int j = 0; j < 4; ++j) {        // B: 1024 16B segments
            int seg = tid * 4 + j;
            int row = seg >> 2;
            int off = (seg & 3) * 8;
            async_copy_b128(Bt + (size_t)(n0 + row) * K + k0 + off,
                            &Bs[buf][row][off]);
        }
    };

    issue_tile(0, 0);

    for (int kt = 0; kt < ksteps; ++kt) {
        int buf = kt & 1;
        __syncthreads();                 // prior readers of buf^1 are done
        if (kt + 1 < ksteps) {
            issue_tile(kt + 1, buf ^ 1); // overlap next copy with this compute
            asm volatile("s_wait_asynccnt 0x6" ::: "memory"); // tile kt complete
        } else {
            asm volatile("s_wait_asynccnt 0x0" ::: "memory");
        }
        __syncthreads();                 // tile kt visible to all waves

        Frag16 af[4];
#pragma unroll
        for (int mf = 0; mf < 4; ++mf) {
            const f16* p = &As[buf][mbase + mf * 16 + nlo][g * 8];
            af[mf].q[0] = *(const u32x4*)p;
            af[mf].q[1] = *(const u32x4*)(p + 16);
        }
#pragma unroll
        for (int nf = 0; nf < 4; ++nf) {
            Frag16 bf;
            const f16* p = &Bs[buf][nbase + nf * 16 + nlo][g * 8];
            bf.q[0] = *(const u32x4*)p;
            bf.q[1] = *(const u32x4*)(p + 16);
#pragma unroll
            for (int mf = 0; mf < 4; ++mf)
                acc[mf][nf] = wmma_f16(af[mf].v, bf.v, acc[mf][nf]);
        }
    }

    // ---------------- epilogue
#pragma unroll
    for (int mf = 0; mf < 4; ++mf)
#pragma unroll
        for (int nf = 0; nf < 4; ++nf)
#pragma unroll
            for (int r = 0; r < 8; ++r) {
                int m = m0 + mbase + mf * 16 + r + 8 * g;
                int n = n0 + nbase + nf * 16 + nlo;
                float val = (acc[mf][nf][r] + bias[n]) * scale;
                if (mode == 0) {
                    int b = m >> 11, s = m & (S_ - 1);
                    int h = n >> 6,  d = n & (HD_ - 1);
                    outH[(((size_t)b * H_ + h) * S_ + s) * HD_ + d] = (f16)val;
                } else if (mode == 2) {
                    int b = m >> 11, s = m & (S_ - 1);
                    int h = n >> 6,  d = n & (HD_ - 1);
                    outH[(((size_t)b * H_ + h) * HD_ + d) * S_ + s] = (f16)val;
                } else if (mode == 3) {
                    outH[(size_t)m * N + n] = (f16)fmaxf(val, 0.f);
                } else { // mode 4
                    outF[(size_t)m * N + n] = val + resid[(size_t)m * N + n];
                }
            }
}

// ---------------------------------------------------------------- flash attn
// One 16-row Q tile per wave; online softmax over 64 key blocks of 32.
// q:  [BH][S][HD] f16 (already scaled by 1/sqrt(HD))
// k:  [BH][S][HD] f16   (serves as Bt for Q*K^T)
// vt: [BH][HD][S] f16   (serves as Bt for P*V)
__global__ __launch_bounds__(256)
void flash_attn_kernel(const f16* __restrict__ q, const f16* __restrict__ k,
                       const f16* __restrict__ vt, const int* __restrict__ mask,
                       f16* __restrict__ attn) {
    __shared__ __align__(16) f16 plds[8][16][40];

    int tid  = threadIdx.x;
    int w    = tid >> 5;
    int lane = tid & 31;
    int g    = lane >> 4;
    int nlo  = lane & 15;

    int gw = blockIdx.x * 8 + w;       // global wave id -> q tile
    int bh = gw >> 7;                  // S/16 = 128 tiles per (b,h)
    int qt = gw & 127;
    int b  = bh >> 4;                  // H = 16
    int s0 = qt * 16;

    // Q fragments (A layout): row = s0 + nlo, two 32-wide K steps over HD
    Frag16 qa[2];
    const f16* qrow = q + ((size_t)bh * S_ + s0 + nlo) * HD_;
#pragma unroll
    for (int ks = 0; ks < 2; ++ks) {
        const f16* p = qrow + ks * 32 + g * 8;
        qa[ks].q[0] = *(const u32x4*)p;
        qa[ks].q[1] = *(const u32x4*)(p + 16);
    }

    v8f o[4];
#pragma unroll
    for (int i = 0; i < 4; ++i) o[i] = {};
    float rm[8], rl[8];
#pragma unroll
    for (int r = 0; r < 8; ++r) { rm[r] = -1e30f; rl[r] = 0.f; }

    const int* mrow = mask + (size_t)b * S_;

    for (int kb = 0; kb < S_ / 32; ++kb) {
        int kbase = kb * 32;

        // ---- scores S = Q K^T  (16 x 32)
        v8f sc[2]; sc[0] = {}; sc[1] = {};
#pragma unroll
        for (int nf = 0; nf < 2; ++nf) {
            const f16* kr = k + ((size_t)bh * S_ + kbase + nf * 16 + nlo) * HD_;
#pragma unroll
            for (int ks = 0; ks < 2; ++ks) {
                Frag16 kf;
                const f16* p = kr + ks * 32 + g * 8;
                kf.q[0] = *(const u32x4*)p;
                kf.q[1] = *(const u32x4*)(p + 16);
                sc[nf] = wmma_f16(qa[ks].v, kf.v, sc[nf]);
            }
        }

        // ---- mask + online softmax
        float sm[2][8], pf[2][8], mx[8];
#pragma unroll
        for (int nf = 0; nf < 2; ++nf) {
            bool keep = mrow[kbase + nf * 16 + nlo] != 0;
#pragma unroll
            for (int r = 0; r < 8; ++r)
                sm[nf][r] = keep ? sc[nf][r] : NEG_;
        }
#pragma unroll
        for (int r = 0; r < 8; ++r) mx[r] = fmaxf(sm[0][r], sm[1][r]);
#pragma unroll
        for (int off = 1; off < 16; off <<= 1)
#pragma unroll
            for (int r = 0; r < 8; ++r)
                mx[r] = fmaxf(mx[r], __shfl_xor(mx[r], off, 32));

        float rs[8];
#pragma unroll
        for (int r = 0; r < 8; ++r) {
            float mnew  = fmaxf(rm[r], mx[r]);
            float alpha = __expf(rm[r] - mnew);
            rm[r] = mnew;
            float acc = 0.f;
#pragma unroll
            for (int nf = 0; nf < 2; ++nf) {
                float p = __expf(sm[nf][r] - mnew);
                pf[nf][r] = p;
                acc += p;
            }
            rs[r] = acc;
            rl[r] *= alpha;
#pragma unroll
            for (int nf2 = 0; nf2 < 4; ++nf2) o[nf2][r] = o[nf2][r] * alpha;
        }
#pragma unroll
        for (int off = 1; off < 16; off <<= 1)
#pragma unroll
            for (int r = 0; r < 8; ++r) rs[r] += __shfl_xor(rs[r], off, 32);
#pragma unroll
        for (int r = 0; r < 8; ++r) rl[r] += rs[r];

        // ---- relayout P (C/D layout -> A layout) through per-wave LDS
#pragma unroll
        for (int nf = 0; nf < 2; ++nf)
#pragma unroll
            for (int r = 0; r < 8; ++r)
                plds[w][r + 8 * g][nf * 16 + nlo] = (f16)pf[nf][r];
        asm volatile("s_wait_dscnt 0" ::: "memory");   // wave-local LDS RAW

        Frag16 pa;
        {
            const f16* p = &plds[w][nlo][g * 8];
            pa.q[0] = *(const u32x4*)p;
            pa.q[1] = *(const u32x4*)(p + 16);
        }

        // ---- O += P V  (V supplied transposed: k-contiguous over keys)
#pragma unroll
        for (int nf2 = 0; nf2 < 4; ++nf2) {
            Frag16 vf;
            const f16* p = vt + ((size_t)bh * HD_ + nf2 * 16 + nlo) * S_
                              + kbase + g * 8;
            vf.q[0] = *(const u32x4*)p;
            vf.q[1] = *(const u32x4*)(p + 16);
            o[nf2] = wmma_f16(pa.v, vf.v, o[nf2]);
        }
        asm volatile("s_wait_dscnt 0" ::: "memory");   // reads done before next store
    }

    // ---- normalize + write attention output rows (row-major [B,S,DM] f16)
    int h = bh & 15;
#pragma unroll
    for (int nf2 = 0; nf2 < 4; ++nf2)
#pragma unroll
        for (int r = 0; r < 8; ++r) {
            int srow = s0 + r + 8 * g;
            int col  = h * HD_ + nf2 * 16 + nlo;
            attn[((size_t)b * S_ + srow) * DM_ + col] = (f16)(o[nf2][r] / rl[r]);
        }
}

// ---------------------------------------------------------------- launcher
extern "C" void kernel_launch(void* const* d_in, const int* in_sizes, int n_in,
                              void* d_out, int out_size, void* d_ws, size_t ws_size,
                              hipStream_t stream) {
    const float* x    = (const float*)d_in[0];
    const int*   mask = (const int*)d_in[1];
    const float* Wq = (const float*)d_in[2];  const float* bq = (const float*)d_in[3];
    const float* Wk = (const float*)d_in[4];  const float* bk = (const float*)d_in[5];
    const float* Wv = (const float*)d_in[6];  const float* bv = (const float*)d_in[7];
    const float* Wo = (const float*)d_in[8];  const float* bo = (const float*)d_in[9];
    const float* W1 = (const float*)d_in[10]; const float* b1 = (const float*)d_in[11];
    const float* W2 = (const float*)d_in[12]; const float* b2 = (const float*)d_in[13];
    const float* alpha1 = (const float*)d_in[14];
    const float* beta1  = (const float*)d_in[15];
    const float* alpha2 = (const float*)d_in[16];
    const float* beta2  = (const float*)d_in[17];
    float* out = (float*)d_out;

    // workspace carve-up
    char* wp = (char*)d_ws;
    auto alloc = [&](size_t bytes) -> void* {
        void* r = (void*)wp;
        wp += (bytes + 255) & ~(size_t)255;
        return r;
    };
    const size_t NT = NTOK_;
    f16* Wqt = (f16*)alloc((size_t)DM_ * DM_ * 2);
    f16* Wkt = (f16*)alloc((size_t)DM_ * DM_ * 2);
    f16* Wvt = (f16*)alloc((size_t)DM_ * DM_ * 2);
    f16* Wot = (f16*)alloc((size_t)DM_ * DM_ * 2);
    f16* W1t = (f16*)alloc((size_t)DM_ * DF_ * 2);
    f16* W2t = (f16*)alloc((size_t)DF_ * DM_ * 2);
    f16* n1  = (f16*)alloc(NT * DM_ * 2);
    f16* qb  = (f16*)alloc(NT * DM_ * 2);
    f16* kb  = (f16*)alloc(NT * DM_ * 2);
    f16* vtb = (f16*)alloc(NT * DM_ * 2);
    f16* at  = (f16*)alloc(NT * DM_ * 2);
    float* hb = (float*)alloc(NT * DM_ * 4);
    f16* n2  = (f16*)alloc(NT * DM_ * 2);
    f16* a1  = (f16*)alloc(NT * DF_ * 2);

    const float qscale = 0.125f;  // 1/sqrt(HD)

    // weight prep
    {
        int g1 = (DM_ * DM_ + 255) / 256;
        int g2 = (DM_ * DF_ + 255) / 256;
        wprep_kernel<<<g1, 256, 0, stream>>>(Wq, Wqt, DM_, DM_);
        wprep_kernel<<<g1, 256, 0, stream>>>(Wk, Wkt, DM_, DM_);
        wprep_kernel<<<g1, 256, 0, stream>>>(Wv, Wvt, DM_, DM_);
        wprep_kernel<<<g1, 256, 0, stream>>>(Wo, Wot, DM_, DM_);
        wprep_kernel<<<g2, 256, 0, stream>>>(W1, W1t, DM_, DF_);  // -> [DF][DM]
        wprep_kernel<<<g2, 256, 0, stream>>>(W2, W2t, DF_, DM_);  // -> [DM][DF]
    }

    // LN1
    layernorm_f16_kernel<<<NTOK_, 256, 0, stream>>>(x, alpha1, beta1, n1);

    // QKV projections (epilogue scatters into attention layouts)
    {
        dim3 grid((NTOK_ / 128) * (DM_ / 256));
        gemm_wmma_kernel<<<grid, 256, 0, stream>>>(n1, Wqt, bq, nullptr, nullptr, qb,
                                                   NTOK_, DM_, DM_, qscale, 0);
        gemm_wmma_kernel<<<grid, 256, 0, stream>>>(n1, Wkt, bk, nullptr, nullptr, kb,
                                                   NTOK_, DM_, DM_, 1.0f, 0);
        gemm_wmma_kernel<<<grid, 256, 0, stream>>>(n1, Wvt, bv, nullptr, nullptr, vtb,
                                                   NTOK_, DM_, DM_, 1.0f, 2);
    }

    // attention
    flash_attn_kernel<<<(B_ * H_ * (S_ / 16)) / 8, 256, 0, stream>>>(qb, kb, vtb,
                                                                     mask, at);

    // output projection + residual -> h (f32)
    gemm_wmma_kernel<<<(NTOK_ / 128) * (DM_ / 256), 256, 0, stream>>>(
        at, Wot, bo, x, hb, nullptr, NTOK_, DM_, DM_, 1.0f, 4);

    // LN2
    layernorm_f16_kernel<<<NTOK_, 256, 0, stream>>>(hb, alpha2, beta2, n2);

    // FFN
    gemm_wmma_kernel<<<(NTOK_ / 128) * (DF_ / 256), 256, 0, stream>>>(
        n2, W1t, b1, nullptr, nullptr, a1, NTOK_, DF_, DM_, 1.0f, 3);
    gemm_wmma_kernel<<<(NTOK_ / 128) * (DM_ / 256), 256, 0, stream>>>(
        a1, W2t, b2, hb, out, nullptr, NTOK_, DM_, DF_, 1.0f, 4);
}
